// Agent_Actor_44186623541380
// MI455X (gfx1250) — matrix operand order repaired and checked by reading, not compiled, not run on hardware
//
#include <hip/hip_runtime.h>

typedef __attribute__((ext_vector_type(16))) _Float16 v16h;
typedef __attribute__((ext_vector_type(8)))  float    v8f;
typedef __attribute__((ext_vector_type(4)))  float    v4f;

#define B_ROWS 131072
#define D_DIM  512
#define A_DIM  6
#define K_OPP  2
#define S_SAMP 20
#define NSTEP  (D_DIM / 32)      // 16 WMMA K-steps
#define WCOLS  524               // D + K*A

// ---------------- CDNA5 async-to-LDS path detection --------------------------
#if defined(__has_builtin)
#if __has_builtin(__builtin_amdgcn_global_load_async_to_lds_b128)
#define HAVE_ASYNC_LDS 1
#endif
#endif

#if defined(__has_builtin) && __has_builtin(__builtin_amdgcn_s_wait_asynccnt)
#define WAIT_ASYNC(n) __builtin_amdgcn_s_wait_asynccnt(n)
#else
#define WAIT_ASYNC(n) asm volatile("s_wait_asynccnt %0" :: "i"(n) : "memory")
#endif

#if defined(__has_builtin) && __has_builtin(__builtin_amdgcn_s_wait_dscnt)
#define WAIT_DS(n) __builtin_amdgcn_s_wait_dscnt(n)
#else
#define WAIT_DS(n) asm volatile("s_wait_dscnt %0" :: "i"(n) : "memory")
#endif

#ifdef HAVE_ASYNC_LDS
// builtin: (int4 AS1* src, int4 AS3* dst, imm offset, imm cpol)
typedef int v4i_b128 __attribute__((vector_size(16)));
typedef __attribute__((address_space(1))) v4i_b128* gas_t;
typedef __attribute__((address_space(3))) v4i_b128* las_t;
static __device__ __forceinline__ gas_t to_as1(const void* p)
{
    return (gas_t)(unsigned long long)p;
}
static __device__ __forceinline__ las_t to_as3(const void* p)
{
    return (las_t)(unsigned)(unsigned long long)p;   // low 32 bits = LDS offset
}
#endif

// ---------------------------------------------------------------------------
// Kernel 0: pack the 18 weight columns (W_opp[0], W_opp[1], Wx) into f16
// WMMA B-fragment layout. wpack[s][t][lane][h]:
//   K  = 32*s + (h&7) + 8*(lane>>4) + 16*(h>>3)   (16-bit 32x16 B layout)
//   N  = t*16 + (lane & 15)
// ---------------------------------------------------------------------------
__global__ void pack_weights(const float* __restrict__ W_opp,
                             const float* __restrict__ W,
                             _Float16* __restrict__ wpack)
{
    int idx = blockIdx.x * blockDim.x + threadIdx.x;   // 0 .. 16383
    if (idx >= NSTEP * 2 * 32 * 16) return;
    int h    = idx & 15;
    int lane = (idx >> 4) & 31;
    int t    = (idx >> 9) & 1;
    int s    = idx >> 10;
    int g    = lane >> 4;
    int col  = t * 16 + (lane & 15);
    int k    = s * 32 + (h & 7) + 8 * g + 16 * (h >> 3);
    float v = 0.0f;
    if (col < 6)        v = W_opp[(0 * A_DIM + col) * D_DIM + k];
    else if (col < 12)  v = W_opp[(1 * A_DIM + (col - 6)) * D_DIM + k];
    else if (col < 18)  v = W[(col - 12) * WCOLS + k];
    wpack[idx] = (_Float16)v;
}

// ---------------------------------------------------------------------------
// splitmix-style hash for deterministic categorical sampling
// ---------------------------------------------------------------------------
__device__ __forceinline__ unsigned int mix32(unsigned int x)
{
    x ^= x >> 16; x *= 0x7feb352du;
    x ^= x >> 15; x *= 0x846ca68bu;
    x ^= x >> 16;
    return x;
}

__device__ __forceinline__ float u01(unsigned int h)
{
    return (float)(h >> 8) * (1.0f / 16777216.0f);
}

// inverse-CDF sample from 6-way dist; returns action, writes chosen prob
__device__ __forceinline__ int sample6(const float* dist, float u, float* pa)
{
    float c = dist[0];
    int   a = 0;
    float p = dist[0];
    #pragma unroll
    for (int i = 1; i < 6; ++i) {
        bool t = (u >= c);
        if (t) { a = i; p = dist[i]; }
        c += dist[i];
    }
    *pa = p;
    return a;
}

// ---------------------------------------------------------------------------
// Kernel 1: fused GEMM (WMMA f16 16x16x32) + softmax/sampling epilogue.
// 128 threads = 4 waves; each wave owns a 16-row tile; block owns 64 rows.
// x streamed global -> LDS with GLOBAL_LOAD_ASYNC_TO_LDS_B128 in a depth-4
// ASYNCcnt pipeline; WMMA B-fragments double-buffered in registers.
// ---------------------------------------------------------------------------
#define XROW 36   // padded row stride (floats) for the LDS x stage (16B-aligned)
#define NBUF 4    // async pipeline depth

__global__ void __launch_bounds__(128)
agent_actor(const float* __restrict__ x,
            const float* __restrict__ b_opp,
            const float* __restrict__ W,
            const float* __restrict__ bvec,
            const int*   __restrict__ seedp,
            const _Float16* __restrict__ wpack,
            float* __restrict__ out)
{
    __shared__ float lds_logits[64][20];   // 18 logit cols per row (padded)
    __shared__ float lds_Wo[72];           // Wo[a][j] = W[a][512 + j], 6x12
    __shared__ float lds_b[6];
    __shared__ float lds_bopp[12];
#ifdef HAVE_ASYNC_LDS
    __shared__ __align__(16) float xstage[4][NBUF][16][XROW];  // [wave][buf][row][col]
#endif

    const int tid  = threadIdx.x;
    const int wave = tid >> 5;
    const int lane = tid & 31;
    const int g    = lane >> 4;           // half-wave selector
    const int m    = lane & 15;           // row within tile / column

    // stage tiny weights once per block
    if (tid < 72) lds_Wo[tid] = W[(tid / 12) * WCOLS + D_DIM + (tid % 12)];
    if (tid < 6)  lds_b[tid] = bvec[tid];
    if (tid >= 96 && tid < 108) lds_bopp[tid - 96] = b_opp[tid - 96];

    const int rowbase = blockIdx.x * 64 + wave * 16;
    const float* xtile = x + (size_t)rowbase * D_DIM;

    v8f acc0 = {};
    v8f acc1 = {};
    const v16h* wp = (const v16h*)wpack;

#ifdef HAVE_ASYNC_LDS
    // ---- async depth-4 staging: 2 KB (16 rows x 32 f32) per step ----
    const int srow = lane >> 3;           // 0..3
    const int scol = (lane & 7) * 4;      // float index of this lane's 16B chunk

    #define STAGE_STEP(s_, buf_)                                                   \
        do {                                                                       \
            const int d0_ = (s_) * 32;                                             \
            _Pragma("unroll")                                                      \
            for (int j = 0; j < 4; ++j) {                                          \
                const int r_ = j * 4 + srow;                                       \
                __builtin_amdgcn_global_load_async_to_lds_b128(                    \
                    to_as1(xtile + (size_t)r_ * D_DIM + d0_ + scol),               \
                    to_as3(&xstage[wave][(buf_)][r_][scol]), 0, 0);                \
            }                                                                      \
        } while (0)

    STAGE_STEP(0, 0);
    STAGE_STEP(1, 1);
    STAGE_STEP(2, 2);

    // B-fragment register double buffer
    v16h b0 = wp[lane];
    v16h b1 = wp[32 + lane];

    #pragma unroll 4
    for (int s = 0; s < NSTEP; ++s) {
        if (s < NSTEP - 3) {
            WAIT_DS(0);                  // reads of the buffer being restaged drained
            STAGE_STEP(s + 3, (s + 3) & (NBUF - 1));
            WAIT_ASYNC(12);              // groups s+1..s+3 outstanding; group s done
        } else if (s == NSTEP - 3) {
            WAIT_ASYNC(8);
        } else if (s == NSTEP - 2) {
            WAIT_ASYNC(4);
        } else {
            WAIT_ASYNC(0);
        }

        // prefetch next step's B-fragments while consuming this step
        const int sn = (s + 1 < NSTEP) ? (s + 1) : s;
        v16h nb0 = wp[(sn * 2 + 0) * 32 + lane];
        v16h nb1 = wp[(sn * 2 + 1) * 32 + lane];

        const float* xb = &xstage[wave][s & (NBUF - 1)][m][0];
        v4f f0 = *(const v4f*)(xb + 8 * g);
        v4f f1 = *(const v4f*)(xb + 8 * g + 4);
        v4f f2 = *(const v4f*)(xb + 16 + 8 * g);
        v4f f3 = *(const v4f*)(xb + 16 + 8 * g + 4);
        v16h a;
        #pragma unroll
        for (int i = 0; i < 4; ++i) {
            a[i]      = (_Float16)f0[i];
            a[4 + i]  = (_Float16)f1[i];
            a[8 + i]  = (_Float16)f2[i];
            a[12 + i] = (_Float16)f3[i];
        }
        acc0 = __builtin_amdgcn_wmma_f32_16x16x32_f16(false, a, false, b0,
                                                      (short)0, acc0, false, false);
        acc1 = __builtin_amdgcn_wmma_f32_16x16x32_f16(false, a, false, b1,
                                                      (short)0, acc1, false, false);
        b0 = nb0;
        b1 = nb1;
    }
    #undef STAGE_STEP
#else
    // ---- fallback: direct nontemporal global loads ----
    const float* xr = xtile + m * D_DIM;
    #pragma unroll 4
    for (int s = 0; s < NSTEP; ++s) {
        const int d0 = s * 32;
        v4f f0 = __builtin_nontemporal_load((const v4f*)(xr + d0 + 8 * g));
        v4f f1 = __builtin_nontemporal_load((const v4f*)(xr + d0 + 8 * g + 4));
        v4f f2 = __builtin_nontemporal_load((const v4f*)(xr + d0 + 16 + 8 * g));
        v4f f3 = __builtin_nontemporal_load((const v4f*)(xr + d0 + 16 + 8 * g + 4));
        v16h a;
        #pragma unroll
        for (int i = 0; i < 4; ++i) {
            a[i]      = (_Float16)f0[i];
            a[4 + i]  = (_Float16)f1[i];
            a[8 + i]  = (_Float16)f2[i];
            a[12 + i] = (_Float16)f3[i];
        }
        v16h b0 = wp[(s * 2 + 0) * 32 + lane];
        v16h b1 = wp[(s * 2 + 1) * 32 + lane];
        acc0 = __builtin_amdgcn_wmma_f32_16x16x32_f16(false, a, false, b0,
                                                      (short)0, acc0, false, false);
        acc1 = __builtin_amdgcn_wmma_f32_16x16x32_f16(false, a, false, b1,
                                                      (short)0, acc1, false, false);
    }
#endif

    // C/D layout: VGPR j, lane -> M = j + 8*g, N = m. Transpose via LDS.
    {
        const int lr = wave * 16 + 8 * g;
        #pragma unroll
        for (int j = 0; j < 8; ++j) lds_logits[lr + j][m] = acc0[j];
        if (m < 2) {
            #pragma unroll
            for (int j = 0; j < 8; ++j) lds_logits[lr + j][16 + m] = acc1[j];
        }
    }
    __syncthreads();

    // ----- per-row epilogue: threads 0..63 each own one row -----
    if (tid < 64) {
        const int r = blockIdx.x * 64 + tid;
        const unsigned int seedv = (unsigned int)seedp[0] * 0x9E3779B9u;
        const unsigned int rowh  = (unsigned int)r * 2654435761u;

        float l0[6], l1[6], xwx[6];
        #pragma unroll
        for (int a = 0; a < 6; ++a) {
            l0[a]  = lds_logits[tid][a]     + lds_bopp[a];
            l1[a]  = lds_logits[tid][6 + a] + lds_bopp[6 + a];
            xwx[a] = lds_logits[tid][12 + a];
        }

        // softmax -> opponent distributions
        float d0v[6], d1v[6];
        {
            float m0 = l0[0], m1 = l1[0];
            #pragma unroll
            for (int a = 1; a < 6; ++a) { m0 = fmaxf(m0, l0[a]); m1 = fmaxf(m1, l1[a]); }
            float s0 = 0.f, s1 = 0.f;
            #pragma unroll
            for (int a = 0; a < 6; ++a) {
                d0v[a] = __expf(l0[a] - m0); s0 += d0v[a];
                d1v[a] = __expf(l1[a] - m1); s1 += d1v[a];
            }
            const float i0 = 1.0f / s0, i1 = 1.0f / s1;
            #pragma unroll
            for (int a = 0; a < 6; ++a) { d0v[a] *= i0; d1v[a] *= i1; }
        }

        // single fused pass over S samples (p-normalization is linear)
        float o[6] = {0.f, 0.f, 0.f, 0.f, 0.f, 0.f};
        float psum = 0.f;
        #pragma unroll
        for (int s = 0; s < S_SAMP; ++s) {
            const float u0 = u01(mix32(seedv ^ rowh ^ (0x632BE5ABu + (unsigned)s * 0x1B873593u)));
            const float u1 = u01(mix32(seedv ^ rowh ^ (0xC2B2AE35u + (unsigned)s * 0x1B873593u)));
            float p0, p1;
            const int a0 = sample6(d0v, u0, &p0);
            const int a1 = sample6(d1v, u1, &p1);
            const float ps = p0 * p1;
            psum += ps;

            float al[6];
            float mx = -1e30f;
            #pragma unroll
            for (int a = 0; a < 6; ++a) {
                al[a] = xwx[a] + lds_Wo[a * 12 + a0] + lds_Wo[a * 12 + 6 + a1] + lds_b[a];
                mx = fmaxf(mx, al[a]);
            }
            float e[6];
            float es = 0.f;
            #pragma unroll
            for (int a = 0; a < 6; ++a) { e[a] = __expf(al[a] - mx); es += e[a]; }
            const float w = ps / es;
            #pragma unroll
            for (int a = 0; a < 6; ++a) o[a] += w * e[a];
        }
        const float inv = 1.0f / psum;
        #pragma unroll
        for (int a = 0; a < 6; ++a) out[(size_t)r * A_DIM + a] = o[a] * inv;
    }
}

// ---------------------------------------------------------------------------
extern "C" void kernel_launch(void* const* d_in, const int* in_sizes, int n_in,
                              void* d_out, int out_size, void* d_ws, size_t ws_size,
                              hipStream_t stream)
{
    const float* x     = (const float*)d_in[0];
    const float* W_opp = (const float*)d_in[1];
    const float* b_opp = (const float*)d_in[2];
    const float* W     = (const float*)d_in[3];
    const float* b     = (const float*)d_in[4];
    const int*   seed  = (const int*)d_in[5];
    float*       outp  = (float*)d_out;
    _Float16*    wpack = (_Float16*)d_ws;   // 32 KB

    pack_weights<<<64, 256, 0, stream>>>(W_opp, W, wpack);
    agent_actor<<<B_ROWS / 64, 128, 0, stream>>>(x, b_opp, W, b, seed, wpack, outp);
}
